// DSTA_52965536694710
// MI455X (gfx1250) — matrix operand order repaired and checked by reading, not compiled, not run on hardware
//
#include <hip/hip_runtime.h>
#include <hip/hip_bf16.h>

typedef __attribute__((ext_vector_type(16))) _Float16 v16h;
typedef __attribute__((ext_vector_type(8)))  float    v8f;

static __device__ __forceinline__ int iclampi(int v, int lo, int hi) {
    return v < lo ? lo : (v > hi ? hi : v);
}
static __device__ __forceinline__ float sigmoidf_(float v) {
    return 1.0f / (1.0f + expf(-v));
}

// ---------------------------------------------------------------------------
// conv1: 1x1 conv 64->16 over 8x256x256 as WMMA GEMM.
// Per wave: 16-pixel x 16-outch tile, K=64 via two v_wmma_f32_16x16x32_f16.
// A: lane L (g=L/16, m=L%16) row=pixel m; halves h: K = 8g+h (h<8), 8g+h+8 (h>=8)
// B: lane n=L%16 col=outch; halves h: K = 8g+h (h<8), 8g+h+8 (h>=8)
// C: vgpr v, lane L: row m=v+8g, col n=L%16
// ---------------------------------------------------------------------------
__global__ __launch_bounds__(256) void k_conv1_wmma(
    const float* __restrict__ x, const float* __restrict__ w1,
    const float* __restrict__ b1, float* __restrict__ out)
{
    __shared__ float sW[1024];   // (16,64)
    __shared__ float sB[16];
    int tid = threadIdx.x;
    for (int i = tid; i < 1024; i += 256) sW[i] = w1[i];
    if (tid < 16) sB[tid] = b1[tid];
    __syncthreads();

    int wave = tid >> 5, lane = tid & 31;
    int g = lane >> 4, m = lane & 15, n = lane & 15;
    int tile = blockIdx.x * 8 + wave;          // 32768 tiles total
    int b    = tile >> 12;                     // tile / 4096 (4096 tiles per image)
    int hw0  = (tile & 4095) << 4;             // pixel base within image
    const size_t xbase = ((size_t)b << 6) << 16;   // b*64*65536

    v8f acc = {};
#pragma unroll
    for (int ks = 0; ks < 2; ++ks) {
        int k0 = ks * 32;
        v16h a, bm;
#pragma unroll
        for (int h = 0; h < 16; ++h) {
            int ca = k0 + 8 * g + h + (h < 8 ? 0 : 8);
            a[h] = (_Float16)x[xbase + ((size_t)ca << 16) + hw0 + m];
            int cb = k0 + 8 * g + h + (h < 8 ? 0 : 8);
            bm[h] = (_Float16)sW[n * 64 + cb];
        }
        acc = __builtin_amdgcn_wmma_f32_16x16x32_f16(false, a, false, bm,
                                                     (short)0, acc, false, false);
    }
#pragma unroll
    for (int v = 0; v < 8; ++v) {
        int mm = v + 8 * g;
        out[((size_t)(b * 16 + n) << 16) + hw0 + mm] = acc[v] + sB[n];
    }
}

// ---------------------------------------------------------------------------
// Implicit-GEMM WMMA 3x3 conv, stride 1, pad 1, Cin=16, Cout multiple of 16.
// K = 144 (tap-major, channel-minor) processed as 5 chunks of 32 (last padded).
// A (lane g,m = pixel): halves h<8 -> tap 2ks,   ch 8g+h
//                       halves h>=8 -> tap 2ks+1, ch 8g+h-8
// B (lane n = outch):   halves h   -> tap 2ks+(h>=8), ch 8g+(h&7) (tap>8 => zero)
// Used for conv_max / conv3 / conv3_ (Cout=16) and mask / mask2 (Cout=432).
// ---------------------------------------------------------------------------
__global__ __launch_bounds__(256) void k_c3x3p1_wmma(
    const float* __restrict__ in, const float* __restrict__ w,
    const float* __restrict__ bias, float* __restrict__ out,
    int B, int H, int W, int Cout, int relu)
{
    __shared__ _Float16 sWh[2304];   // 16 outch x (16 ch x 9 taps)
    __shared__ float sB[16];
    const int tid = threadIdx.x;
    const int o_base = blockIdx.y * 16;
    for (int i = tid; i < 2304; i += 256)
        sWh[i] = (_Float16)w[(size_t)o_base * 144 + i];
    if (tid < 16) sB[tid] = bias[o_base + tid];
    __syncthreads();

    const int HW = H * W;
    const int total = B * HW;
    const int tiles = (total + 15) >> 4;
    int wave = tid >> 5, lane = tid & 31;
    int tile = blockIdx.x * 8 + wave;
    if (tile >= tiles) return;                 // wave-uniform exit (after barrier)
    int g = lane >> 4, m = lane & 15, n = lane & 15;
    int p  = tile * 16 + m;
    bool pv = p < total;
    int pp = pv ? p : 0;
    int b  = pp / HW, rem = pp % HW;
    int r  = rem / W, cx = rem % W;
    const float* ib = in + (size_t)b * 16 * HW;

    v8f acc = {};
    for (int ks = 0; ks < 5; ++ks) {
        v16h a, bm;
#pragma unroll
        for (int half = 0; half < 2; ++half) {
            int k  = 2 * ks + half;            // kernel tap for this half-group
            int ky = k / 3, kx = k - 3 * (k / 3);
            int iy = r - 1 + ky, ix = cx - 1 + kx;
            bool iv = pv && (k < 9) &&
                      ((unsigned)iy < (unsigned)H) && ((unsigned)ix < (unsigned)W);
#pragma unroll
            for (int hh = 0; hh < 8; ++hh) {
                int ci = 8 * g + hh;
                float v = iv ? ib[(size_t)ci * HW + iy * W + ix] : 0.0f;
                a[half * 8 + hh] = (_Float16)v;
            }
        }
        int kb = 2 * ks;
        bool bv = kb < 9;
#pragma unroll
        for (int h = 0; h < 16; ++h)
            bm[h] = (kb + (h >> 3) < 9) ? sWh[n * 144 + (8 * g + (h & 7)) * 9 + kb + (h >> 3)] : (_Float16)0.0f;
        acc = __builtin_amdgcn_wmma_f32_16x16x32_f16(false, a, false, bm,
                                                     (short)0, acc, false, false);
    }
#pragma unroll
    for (int v = 0; v < 8; ++v) {
        int pe = tile * 16 + v + 8 * g;
        if (pe >= total) continue;
        int be = pe / HW, re = pe - be * HW;
        float val = acc[v] + sB[n];
        if (relu) val = fmaxf(val, 0.0f);
        out[(size_t)(be * Cout + o_base + n) * HW + re] = val;
    }
}

// ---------------------------------------------------------------------------
// Generic direct 3x3 conv (strided stages conv2 / down2 only)
// ---------------------------------------------------------------------------
__global__ __launch_bounds__(256) void k_conv3x3(
    const float* __restrict__ in, const float* __restrict__ w,
    const float* __restrict__ bias, float* __restrict__ out,
    int B, int Cin, int H, int W, int Cout, int Ho, int Wo,
    int stride, int pad, int relu)
{
    int idx = blockIdx.x * blockDim.x + threadIdx.x;
    int total = B * Cout * Ho * Wo;
    if (idx >= total) return;
    int xo = idx % Wo; int t = idx / Wo;
    int yo = t % Ho;  t /= Ho;
    int o  = t % Cout; int b = t / Cout;

    float acc = bias[o];
    const float* wp = w + (size_t)o * Cin * 9;
    const float* ip = in + (size_t)b * Cin * H * W;
    for (int c = 0; c < Cin; ++c) {
        const float* cp = ip + (size_t)c * H * W;
        const float* wc = wp + c * 9;
#pragma unroll
        for (int ky = 0; ky < 3; ++ky) {
            int iy = yo * stride - pad + ky;
            if ((unsigned)iy >= (unsigned)H) continue;
#pragma unroll
            for (int kx = 0; kx < 3; ++kx) {
                int ix = xo * stride - pad + kx;
                if ((unsigned)ix >= (unsigned)W) continue;
                acc += cp[iy * W + ix] * wc[ky * 3 + kx];
            }
        }
    }
    if (relu) acc = fmaxf(acc, 0.0f);
    out[idx] = acc;
}

// ---------------------------------------------------------------------------
// 7x7 stride-3 VALID max pool: 127x127 -> 41x41, 128 planes
// ---------------------------------------------------------------------------
__global__ __launch_bounds__(256) void k_maxpool(const float* __restrict__ in,
                                                 float* __restrict__ out)
{
    int idx = blockIdx.x * blockDim.x + threadIdx.x;
    if (idx >= 128 * 41 * 41) return;
    int xo = idx % 41, yo = (idx / 41) % 41, p = idx / 1681;
    const float* ip = in + (size_t)p * 127 * 127;
    float mx = -3.4e38f;
#pragma unroll
    for (int wy = 0; wy < 7; ++wy)
#pragma unroll
        for (int wx = 0; wx < 7; ++wx)
            mx = fmaxf(mx, ip[(yo * 3 + wy) * 127 + (xo * 3 + wx)]);
    out[idx] = mx;
}

// ---------------------------------------------------------------------------
// off_msk += bilinear_resize(off_mask2, 21->41); sigmoid on channels >= 288
// ---------------------------------------------------------------------------
__global__ __launch_bounds__(256) void k_offmask(float* __restrict__ om,
                                                 const float* __restrict__ om2)
{
    int idx = blockIdx.x * blockDim.x + threadIdx.x;
    if (idx >= 8 * 432 * 1681) return;
    int xo = idx % 41, yo = (idx / 41) % 41;
    int ch = (idx / 1681) % 432, b = idx / (1681 * 432);
    const float* p2 = om2 + (size_t)(b * 432 + ch) * 441;

    float sy = (yo + 0.5f) * (21.0f / 41.0f) - 0.5f;
    float sx = (xo + 0.5f) * (21.0f / 41.0f) - 0.5f;
    float y0f = floorf(sy), x0f = floorf(sx);
    float fy = sy - y0f, fx = sx - x0f;
    int y0 = iclampi((int)y0f, 0, 20), y1 = iclampi((int)y0f + 1, 0, 20);
    int x0 = iclampi((int)x0f, 0, 20), x1 = iclampi((int)x0f + 1, 0, 20);
    float up = (1 - fy) * ((1 - fx) * p2[y0 * 21 + x0] + fx * p2[y0 * 21 + x1])
             +      fy  * ((1 - fx) * p2[y1 * 21 + x0] + fx * p2[y1 * 21 + x1]);
    float v = om[idx] + up;
    if (ch >= 288) v = sigmoidf_(v);
    om[idx] = v;
}

// ---------------------------------------------------------------------------
// DCNv2 (dg=16, cg=1, 3x3, pad1): thread per (b, o, y, x)
// ---------------------------------------------------------------------------
__global__ __launch_bounds__(256) void k_dcn(
    const float* __restrict__ vm, const float* __restrict__ om,
    const float* __restrict__ wk, const float* __restrict__ bk,
    float* __restrict__ out)
{
    int idx = blockIdx.x * blockDim.x + threadIdx.x;
    if (idx >= 8 * 16 * 1681) return;
    int xo = idx % 41, yo = (idx / 41) % 41;
    int o  = (idx / 1681) % 16, b = idx / (1681 * 16);
    int pidx = yo * 41 + xo;
    const float* omb = om + (size_t)b * 432 * 1681;

    float acc = bk[o];
    for (int g = 0; g < 16; ++g) {
        const float* vp = vm + (size_t)(b * 16 + g) * 1681;
#pragma unroll
        for (int k = 0; k < 9; ++k) {
            int ky = k / 3, kx = k % 3;
            float dy = omb[(size_t)((g * 9 + k) * 2 + 0) * 1681 + pidx];
            float dx = omb[(size_t)((g * 9 + k) * 2 + 1) * 1681 + pidx];
            float mv = omb[(size_t)(288 + g * 9 + k) * 1681 + pidx];
            float py = dy + (float)(yo - 1 + ky);
            float px = dx + (float)(xo - 1 + kx);
            float y0f = floorf(py), x0f = floorf(px);
            float ly = py - y0f, lx = px - x0f;
            int y0 = (int)y0f, x0 = (int)x0f;
            auto C = [&](int yi, int xi) -> float {
                if (yi < 0 || yi > 40 || xi < 0 || xi > 40) return 0.0f;
                return vp[yi * 41 + xi];
            };
            float s = C(y0, x0)     * (1 - ly) * (1 - lx)
                    + C(y0, x0 + 1) * (1 - ly) * lx
                    + C(y0 + 1, x0) * ly * (1 - lx)
                    + C(y0 + 1, x0 + 1) * ly * lx;
            acc += s * mv * wk[(o * 16 + g) * 9 + k];
        }
    }
    out[idx] = fmaxf(acc, 0.0f);
}

// spatial mean over 41x41, one block per (b,c) plane
__global__ __launch_bounds__(256) void k_mean(const float* __restrict__ in,
                                              float* __restrict__ out)
{
    __shared__ float red[256];
    int p = blockIdx.x;
    const float* ip = in + (size_t)p * 1681;
    float s = 0.0f;
    for (int i = threadIdx.x; i < 1681; i += 256) s += ip[i];
    red[threadIdx.x] = s;
    __syncthreads();
    for (int st = 128; st > 0; st >>= 1) {
        if (threadIdx.x < st) red[threadIdx.x] += red[threadIdx.x + st];
        __syncthreads();
    }
    if (threadIdx.x == 0) out[p] = red[0] * (1.0f / 1681.0f);
}

// squeeze-excite: du1 (32,16) relu, du2 (64,32) sigmoid; one block
__global__ __launch_bounds__(256) void k_se(
    const float* __restrict__ mean, const float* __restrict__ w1,
    const float* __restrict__ b1, const float* __restrict__ w2,
    const float* __restrict__ b2, float* __restrict__ y)
{
    __shared__ float t[256];   // 8 x 32
    int tid = threadIdx.x;
    int b = tid >> 5, j = tid & 31;
    float acc = b1[j];
    for (int c = 0; c < 16; ++c) acc += w1[j * 16 + c] * mean[b * 16 + c];
    t[b * 32 + j] = fmaxf(acc, 0.0f);
    __syncthreads();
#pragma unroll
    for (int r = 0; r < 2; ++r) {
        int oid = tid + r * 256;          // 512 outputs
        int bb = oid >> 6, o = oid & 63;
        float a2 = b2[o];
        for (int jj = 0; jj < 32; ++jj) a2 += w2[o * 32 + jj] * t[bb * 32 + jj];
        y[oid] = sigmoidf_(a2);
    }
}

// ---------------------------------------------------------------------------
// Final fused kernel: per 16-pixel tile of 256x256:
//  t = bilinear_up(dcn, 41->256) + conv_f(c1_)      (16 channels)
//  c4 = W4(64x16) @ t via 4x v_wmma_f32_16x16x32_f16
//  out = x * sigmoid(c4 + b4) * y[b]
// ---------------------------------------------------------------------------
__global__ __launch_bounds__(256) void k_final(
    const float* __restrict__ x, const float* __restrict__ c1f,
    const float* __restrict__ dcn, const float* __restrict__ wf,
    const float* __restrict__ bfv, const float* __restrict__ w4,
    const float* __restrict__ b4, const float* __restrict__ yv,
    float* __restrict__ out)
{
    __shared__ float sWF[256], sBF[16], sW4[1024], sB4[64], sY[64];
    int tid = threadIdx.x;
    int tile0 = blockIdx.x * 8;
    int b = tile0 >> 12;                       // uniform per block
    for (int i = tid; i < 1024; i += 256) sW4[i] = w4[i];
    if (tid < 256) sWF[tid] = wf[tid];
    if (tid < 16) sBF[tid] = bfv[tid];
    if (tid < 64) { sB4[tid] = b4[tid]; sY[tid] = yv[b * 64 + tid]; }
    __syncthreads();

    int wave = tid >> 5, lane = tid & 31;
    int g = lane >> 4, m = lane & 15, n = lane & 15;
    int tile = tile0 + wave;
    int pix0 = (tile & 4095) << 4;             // within-image pixel base
    int py = pix0 >> 8, px0 = pix0 & 255;
    int pxm = px0 + m;

    // bilinear 41->256 taps for pixel (py, pxm)
    float sy = (py + 0.5f) * (41.0f / 256.0f) - 0.5f;
    float sx = (pxm + 0.5f) * (41.0f / 256.0f) - 0.5f;
    float y0f = floorf(sy), x0f = floorf(sx);
    float fy = sy - y0f, fx = sx - x0f;
    int y0 = iclampi((int)y0f, 0, 40), y1 = iclampi((int)y0f + 1, 0, 40);
    int x0 = iclampi((int)x0f, 0, 40), x1 = iclampi((int)x0f + 1, 0, 40);
    int i00 = y0 * 41 + x0, i01 = y0 * 41 + x1, i10 = y1 * 41 + x0, i11 = y1 * 41 + x1;
    float w00 = (1 - fy) * (1 - fx), w01 = (1 - fy) * fx;
    float w10 = fy * (1 - fx),       w11 = fy * fx;

    // c1_ pixel vector (16 channels)
    float c1p[16];
    size_t pidx = (size_t)py * 256 + pxm;
#pragma unroll
    for (int j = 0; j < 16; ++j)
        c1p[j] = c1f[((size_t)(b * 16 + j) << 16) + pidx];

    // A fragment: lane group g computes channels [8g, 8g+8)
    v16h a;
#pragma unroll
    for (int h = 0; h < 8; ++h) {
        int c = 8 * g + h;
        const float* dp = dcn + (size_t)(b * 16 + c) * 1681;
        float up = w00 * dp[i00] + w01 * dp[i01] + w10 * dp[i10] + w11 * dp[i11];
        float cf = sBF[c];
#pragma unroll
        for (int j = 0; j < 16; ++j) cf += sWF[c * 16 + j] * c1p[j];
        a[h]     = (_Float16)(up + cf);
        a[h + 8] = (_Float16)0.0f;
    }

    const size_t outbase = (size_t)b << 22;    // b*64*65536
#pragma unroll
    for (int j = 0; j < 4; ++j) {
        v16h bm;
#pragma unroll
        for (int h = 0; h < 16; ++h)
            bm[h] = (h < 8) ? (_Float16)sW4[(16 * j + n) * 16 + 8 * g + h] : (_Float16)0.0f;
        v8f c = {};
        c = __builtin_amdgcn_wmma_f32_16x16x32_f16(false, a, false, bm,
                                                   (short)0, c, false, false);
#pragma unroll
        for (int v = 0; v < 8; ++v) {
            int mm = v + 8 * g;
            int o  = 16 * j + n;
            float mg = sigmoidf_(c[v] + sB4[o]);
            size_t xi = outbase + ((size_t)o << 16) + (size_t)py * 256 + (px0 + mm);
            out[xi] = x[xi] * mg * sY[o];
        }
    }
}

// ---------------------------------------------------------------------------
extern "C" void kernel_launch(void* const* d_in, const int* in_sizes, int n_in,
                              void* d_out, int out_size, void* d_ws, size_t ws_size,
                              hipStream_t stream) {
    const float* x        = (const float*)d_in[0];
    const float* conv1_w  = (const float*)d_in[1];
    const float* conv1_b  = (const float*)d_in[2];
    const float* convf_w  = (const float*)d_in[3];
    const float* convf_b  = (const float*)d_in[4];
    const float* convm_w  = (const float*)d_in[5];
    const float* convm_b  = (const float*)d_in[6];
    const float* conv2_w  = (const float*)d_in[7];
    const float* conv2_b  = (const float*)d_in[8];
    const float* conv3_w  = (const float*)d_in[9];
    const float* conv3_b  = (const float*)d_in[10];
    const float* conv3b_w = (const float*)d_in[11];
    const float* conv3b_b = (const float*)d_in[12];
    const float* conv4_w  = (const float*)d_in[13];
    const float* conv4_b  = (const float*)d_in[14];
    const float* dcn_w    = (const float*)d_in[15];
    const float* dcn_b    = (const float*)d_in[16];
    const float* mask_w   = (const float*)d_in[17];
    const float* mask_b   = (const float*)d_in[18];
    const float* down2_w  = (const float*)d_in[19];
    const float* down2_b  = (const float*)d_in[20];
    const float* mask2_w  = (const float*)d_in[21];
    const float* mask2_b  = (const float*)d_in[22];
    const float* du1_w    = (const float*)d_in[23];
    const float* du1_b    = (const float*)d_in[24];
    const float* du2_w    = (const float*)d_in[25];
    const float* du2_b    = (const float*)d_in[26];
    float* out = (float*)d_out;

    // workspace carving (floats)
    float* ws = (float*)d_ws;
    size_t off = 0;
    float* c1f  = ws + off; off += (size_t)8 * 16 * 65536;      // 8,388,608
    float* c1   = ws + off; off += (size_t)8 * 16 * 127 * 127;  // 2,064,512
    float* vmx  = ws + off; off += (size_t)8 * 16 * 1681;
    float* vrg  = ws + off; off += (size_t)8 * 16 * 1681;
    float* c3a  = ws + off; off += (size_t)8 * 16 * 1681;
    float* c3b  = ws + off; off += (size_t)8 * 16 * 1681;
    float* dc3  = ws + off; off += (size_t)8 * 16 * 441;
    float* om2  = ws + off; off += (size_t)8 * 432 * 441;
    float* om   = ws + off; off += (size_t)8 * 432 * 1681;
    float* dcnb = ws + off; off += (size_t)8 * 16 * 1681;
    float* mnb  = ws + off; off += 128;
    float* yvb  = ws + off; off += 512;

    auto cdiv = [](int a, int b) { return (a + b - 1) / b; };
    const int gx41 = cdiv(cdiv(8 * 41 * 41, 16), 8);   // 106 blocks (841 tiles)
    const int gx21 = cdiv(cdiv(8 * 21 * 21, 16), 8);   // 28 blocks (221 tiles)

    // 1. conv1 1x1 64->16 (WMMA)
    k_conv1_wmma<<<4096, 256, 0, stream>>>(x, conv1_w, conv1_b, c1f);
    // 2. conv2 3x3 s2 p0: 256 -> 127 (direct; strided)
    k_conv3x3<<<cdiv(8 * 16 * 127 * 127, 256), 256, 0, stream>>>(
        c1f, conv2_w, conv2_b, c1, 8, 16, 256, 256, 16, 127, 127, 2, 0, 0);
    // 3. maxpool 7x7 s3: 127 -> 41
    k_maxpool<<<cdiv(128 * 1681, 256), 256, 0, stream>>>(c1, vmx);
    // 4. conv_max 3x3 p1 relu (WMMA implicit GEMM)
    k_c3x3p1_wmma<<<dim3(gx41, 1), 256, 0, stream>>>(
        vmx, convm_w, convm_b, vrg, 8, 41, 41, 16, 1);
    // 5. conv3 3x3 p1 relu (WMMA)
    k_c3x3p1_wmma<<<dim3(gx41, 1), 256, 0, stream>>>(
        vrg, conv3_w, conv3_b, c3a, 8, 41, 41, 16, 1);
    // 6. conv3_ 3x3 p1 relu (WMMA)
    k_c3x3p1_wmma<<<dim3(gx41, 1), 256, 0, stream>>>(
        c3a, conv3b_w, conv3b_b, c3b, 8, 41, 41, 16, 1);
    // 7. down2 3x3 s2 p1 relu: 41 -> 21 (direct; strided)
    k_conv3x3<<<cdiv(8 * 16 * 441, 256), 256, 0, stream>>>(
        c3b, down2_w, down2_b, dc3, 8, 16, 41, 41, 16, 21, 21, 2, 1, 1);
    // 8. mask2 conv 16->432 @21x21 (WMMA, gy=27)
    k_c3x3p1_wmma<<<dim3(gx21, 27), 256, 0, stream>>>(
        dc3, mask2_w, mask2_b, om2, 8, 21, 21, 432, 0);
    // 9. mask conv 16->432 @41x41 (WMMA, gy=27) -- largest-FLOP stage
    k_c3x3p1_wmma<<<dim3(gx41, 27), 256, 0, stream>>>(
        c3b, mask_w, mask_b, om, 8, 41, 41, 432, 0);
    // 10. off_msk += resize(off_mask2); sigmoid mask tail
    k_offmask<<<cdiv(8 * 432 * 1681, 256), 256, 0, stream>>>(om, om2);
    // 11. modulated deformable conv + relu
    k_dcn<<<cdiv(8 * 16 * 1681, 256), 256, 0, stream>>>(vmx, om, dcn_w, dcn_b, dcnb);
    // 12. spatial mean
    k_mean<<<128, 256, 0, stream>>>(dcnb, mnb);
    // 13. squeeze-excite
    k_se<<<1, 256, 0, stream>>>(mnb, du1_w, du1_b, du2_w, du2_b, yvb);
    // 14. fused upsample + conv_f + conv4 (WMMA) + gating
    k_final<<<4096, 256, 0, stream>>>(x, c1f, dcnb, convf_w, convf_b,
                                      conv4_w, conv4_b, yvb, out);
}
